// PatchedAttention_58815282151822
// MI455X (gfx1250) — matrix operand order repaired and checked by reading, not compiled, not run on hardware
//
#include <hip/hip_runtime.h>
#include <hip/hip_bf16.h>

// ---------------------------------------------------------------------------
// Patched multi-head attention with relative-position bias for gfx1250.
// One workgroup (256 threads = 8 wave32 waves) handles one 64-token patch.
// Wave w owns head w. All GEMMs use v_wmma_f32_16x16x32_f16.
// ---------------------------------------------------------------------------

#define DMODEL 256
#define PATCH  64
#define NHEAD  8
#define DHEAD  32
#define NPATCH 1024   // B*T/PATCH = 16*4096/64

typedef _Float16 v16h __attribute__((ext_vector_type(16)));
typedef _Float16 v8h  __attribute__((ext_vector_type(8)));
typedef _Float16 v4h  __attribute__((ext_vector_type(4)));
typedef float    v8f  __attribute__((ext_vector_type(8)));

// LDS layout (bytes). Phase-overlapped regions:
//   region A  [0, 33792)      : Xh [64][264] f16   (phase 1-2)
//                               pairIdx[64][64] u16 + rpe[64][8] f32 +
//                               etaphi[128] f32 + per-wave prob tiles (phase 3-4)
//   region W  [33792, 54272)  : Wt [256][40] f16   (phase 2 and 5)
//   region Q  [54272, 88064)  : Qh [64][264] f16   (2-3)  /  Attn (4-5)
//   region K  [88064, 121856) : Kh [64][264] f16   (2-3)
//   region V  [121856,154624) : Vt [256][64] f16   (2-4)
#define OFF_XH   0
#define OFF_WT   33792
#define OFF_Q    54272
#define OFF_K    88064
#define OFF_VT   121856
#define SMEM_TOT 154624
#define XS 264   // row stride (halves) for Xh/Qh/Kh/Attn
#define WS 40    // row stride (halves) for Wt (transposed weight tile)
#define PRS 72   // row stride (halves) of per-wave prob tile
#define OFF_PAIR (OFF_XH)
#define OFF_RPE  (OFF_XH + 8192)
#define OFF_ETA  (OFF_XH + 10240)
#define OFF_PROB (OFF_XH + 10752)   // 8 waves * 16*72*2 = 18432 bytes

__device__ __forceinline__ v8f vzero8() {
  v8f z;
#pragma unroll
  for (int e = 0; e < 8; ++e) z[e] = 0.0f;
  return z;
}

__device__ __forceinline__ v8f wmma_f16(v16h a, v16h b, v8f c) {
  return __builtin_amdgcn_wmma_f32_16x16x32_f16(false, a, false, b,
                                                (short)0, c, false, false);
}

// A-frag (16x32 f16, row-major source with leading dim ldh):
// lane L holds row row0+(L&15); VGPR0-3 = K[k0+8g .. +7], VGPR4-7 = K[k0+16+8g .. +7]
__device__ __forceinline__ v16h load_a_frag(const _Float16* base, int row0,
                                            int ldh, int k0, int lane) {
  const int g = lane >> 4, m = lane & 15;
  const _Float16* p = base + (row0 + m) * ldh + k0 + 8 * g;
  v8h lo = *(const v8h*)p;
  v8h hi = *(const v8h*)(p + 16);
  return __builtin_shufflevector(lo, hi, 0, 1, 2, 3, 4, 5, 6, 7,
                                 8, 9, 10, 11, 12, 13, 14, 15);
}

// B-frag (32x16 f16) from a [N][K] (transposed) source with leading dim ldh:
// lane L holds column col0+(L&15); VGPR0-3 = K[k0+8g .. +7], VGPR4-7 = K[k0+16+8g .. +7]
__device__ __forceinline__ v16h load_b_frag(const _Float16* base, int col0,
                                            int ldh, int k0, int lane) {
  const int g = lane >> 4, n = lane & 15;
  const _Float16* p = base + (col0 + n) * ldh + k0 + 8 * g;
  v8h lo = *(const v8h*)p;
  v8h hi = *(const v8h*)(p + 16);
  return __builtin_shufflevector(lo, hi, 0, 1, 2, 3, 4, 5, 6, 7,
                                 8, 9, 10, 11, 12, 13, 14, 15);
}

// Stage a transposed 32x256 fp32 weight tile into Wt[outcol][kk] as f16.
// float4 (b128) global loads along the contiguous out-column dimension.
__device__ __forceinline__ void stage_wtile(const float* __restrict__ Wsrc,
                                            int k0, _Float16* Wt, int tid) {
  for (int e = tid; e < (32 * 256) / 4; e += 256) {
    int e4 = e << 2;
    int kk = e4 >> 8;
    int n0 = e4 & 255;
    const float4 v = *(const float4*)(Wsrc + (k0 + kk) * DMODEL + n0);
    Wt[(n0 + 0) * WS + kk] = (_Float16)v.x;
    Wt[(n0 + 1) * WS + kk] = (_Float16)v.y;
    Wt[(n0 + 2) * WS + kk] = (_Float16)v.z;
    Wt[(n0 + 3) * WS + kk] = (_Float16)v.w;
  }
}

// One 64x256 @ 256x256 projection pass. MODE: 0 -> Qh, 1 -> Kh, 2 -> Vt.
// Templated so each instantiation sees the kernel-arg pointer directly
// (keeps the loads on the GLOBAL path instead of FLAT).
template <int MODE>
__device__ __forceinline__ void proj_pass(
    const float* __restrict__ W, const float* __restrict__ bvec,
    const _Float16* Xh, _Float16* Wt, _Float16* dst,
    int tid, int lane, int w, int g, int n) {
  v8f acc[4][2];
#pragma unroll
  for (int i = 0; i < 4; ++i) {
    acc[i][0] = vzero8();
    acc[i][1] = vzero8();
  }
  for (int k0 = 0; k0 < DMODEL; k0 += 32) {
    stage_wtile(W, k0, Wt, tid);
    if (k0 + 32 < DMODEL)  // pull next tile toward L2/L0 while we compute
      __builtin_prefetch(W + (k0 + 32) * DMODEL + tid * 32, 0, 1);
    __syncthreads();
    v16h bf0 = load_b_frag(Wt, 32 * w, WS, 0, lane);
    v16h bf1 = load_b_frag(Wt, 32 * w + 16, WS, 0, lane);
#pragma unroll
    for (int i = 0; i < 4; ++i) {
      v16h af = load_a_frag(Xh, 16 * i, XS, k0, lane);
      acc[i][0] = wmma_f16(af, bf0, acc[i][0]);
      acc[i][1] = wmma_f16(af, bf1, acc[i][1]);
    }
    __syncthreads();
  }
  // bias + store to LDS (Q,K row-major; V transposed [dim][token])
#pragma unroll
  for (int jj = 0; jj < 2; ++jj) {
    int col = 32 * w + 16 * jj + n;
    float bias = bvec[col];
#pragma unroll
    for (int i = 0; i < 4; ++i) {
#pragma unroll
      for (int m = 0; m < 8; ++m) {
        int token = 16 * i + m + 8 * g;
        _Float16 hv = (_Float16)(acc[i][jj][m] + bias);
        if (MODE == 2) dst[col * 64 + token] = hv;
        else           dst[token * XS + col] = hv;
      }
    }
  }
}

// ---------------------------------------------------------------------------
// Kernel 1: global eta range = max over all patches of (max eta - min eta)
// ---------------------------------------------------------------------------
__global__ __launch_bounds__(64) void eta_range_kernel(const float* __restrict__ coords,
                                                       unsigned int* __restrict__ range_bits) {
  __shared__ float se[PATCH];
  const int t = threadIdx.x;
  const long base = (long)blockIdx.x * PATCH;
  se[t] = coords[(base + t) * 2 + 0];
  __syncthreads();
  if (t == 0) {
    float mx = -1e30f, mn = 1e30f;
    for (int i = 0; i < PATCH; ++i) {
      mx = fmaxf(mx, se[i]);
      mn = fminf(mn, se[i]);
    }
    float r = mx - mn;  // >= 0, so float bits compare monotonically
    atomicMax(range_bits, __float_as_uint(r));
  }
}

// ---------------------------------------------------------------------------
// Kernel 2: full patched attention
// ---------------------------------------------------------------------------
__global__ __launch_bounds__(256) void patched_attn_kernel(
    const float* __restrict__ x, const float* __restrict__ coords,
    const float* __restrict__ wq, const float* __restrict__ bq,
    const float* __restrict__ wk, const float* __restrict__ bk,
    const float* __restrict__ wv, const float* __restrict__ bv,
    const float* __restrict__ wo, const float* __restrict__ bo,
    const float* __restrict__ rpe_table, float* __restrict__ out,
    const float* __restrict__ range_ptr) {
  __shared__ __align__(16) unsigned char smem[SMEM_TOT];

  const int tid  = threadIdx.x;
  const int lane = tid & 31;
  const int w    = tid >> 5;        // wave id == head id
  const int g    = lane >> 4;
  const int n    = lane & 15;
  const long pbase = (long)blockIdx.x * PATCH;   // global token base

  _Float16* Xh   = (_Float16*)(smem + OFF_XH);   // [64][XS]
  _Float16* Wt   = (_Float16*)(smem + OFF_WT);   // [256][WS] transposed tile
  _Float16* Qh   = (_Float16*)(smem + OFF_Q);    // [64][XS]
  _Float16* Kh   = (_Float16*)(smem + OFF_K);    // [64][XS]
  _Float16* Vt   = (_Float16*)(smem + OFF_VT);   // [256][64] (dim-major)
  _Float16* Attn = Qh;                           // reuse after phase 3
  unsigned short* pairIdx = (unsigned short*)(smem + OFF_PAIR);  // [64][64]
  float* rpeS = (float*)(smem + OFF_RPE);        // [64][8]
  float* etaS = (float*)(smem + OFF_ETA);        // eta[64], phi[64]
  _Float16* probW = (_Float16*)(smem + OFF_PROB) + w * (16 * PRS);

  // ---------------- Phase 1: stage x patch as f16 (b128 loads, b64 stores) --
  for (int e = tid; e < (PATCH * DMODEL) / 4; e += 256) {
    int e4 = e << 2;
    int t = e4 >> 8, d = e4 & 255;
    const float4 v = *(const float4*)(x + (pbase + t) * DMODEL + d);
    v4h h;
    h[0] = (_Float16)v.x; h[1] = (_Float16)v.y;
    h[2] = (_Float16)v.z; h[3] = (_Float16)v.w;
    *(v4h*)(Xh + t * XS + d) = h;
  }
  __syncthreads();

  // ---------------- Phase 2: Q/K/V projections ----------------
  proj_pass<0>(wq, bq, Xh, Wt, Qh, tid, lane, w, g, n);
  proj_pass<1>(wk, bk, Xh, Wt, Kh, tid, lane, w, g, n);
  proj_pass<2>(wv, bv, Xh, Wt, Vt, tid, lane, w, g, n);
  __syncthreads();   // region A now free; Q/K/Vt ready

  // ---------------- Phase 3 prep: coords, rpe table, pair indices ----------
  for (int e = tid; e < 2 * PATCH; e += 256) {
    int t = e & 63, c = e >> 6;
    etaS[c * 64 + t] = coords[(pbase + t) * 2 + c];
  }
  for (int e = tid; e < 64 * NHEAD; e += 256) rpeS[e] = rpe_table[e];
  __syncthreads();

  const float PI_F = 3.14159265358979323846f;
  const float eta_range = fmaxf(*range_ptr, 1e-6f);
  for (int p = tid; p < 64 * 64; p += 256) {
    int t = p >> 6, T = p & 63;
    float re = etaS[t] - etaS[T];
    float rp = etaS[64 + t] - etaS[64 + T];
    rp = fmodf(rp + PI_F, 2.0f * PI_F);
    if (rp < 0.0f) rp += 2.0f * PI_F;
    rp -= PI_F;
    int eb = (int)(re / eta_range * 16.0f);
    eb = min(15, max(-16, eb));
    int pb = (int)(rp / PI_F * 16.0f);
    pb = min(15, max(-16, pb));
    int ei = eb + 16;        // 0..31
    int pi = pb + 48;        // 32..63
    pairIdx[p] = (unsigned short)(ei | (pi << 8));
  }
  __syncthreads();

  // Load Q A-frags, K B-frags, V B-frags for head w into registers
  v16h aq[4], bkf[4], bvf[2][2];
#pragma unroll
  for (int i = 0; i < 4; ++i) aq[i] = load_a_frag(Qh, 16 * i, XS, w * DHEAD, lane);
#pragma unroll
  for (int j = 0; j < 4; ++j) bkf[j] = load_b_frag(Kh, 16 * j, XS, w * DHEAD, lane);
#pragma unroll
  for (int kk = 0; kk < 2; ++kk)
#pragma unroll
    for (int jj = 0; jj < 2; ++jj)
      bvf[kk][jj] = load_b_frag(Vt, w * DHEAD + 16 * jj, 64, 32 * kk, lane);
  __syncthreads();   // all waves done reading Qh -> safe to overwrite as Attn

  // ---------------- Phase 3/4: scores + bias + softmax + P@V per row block --
  const float scale = 0.17677669529663687f;  // 1/sqrt(32)
  for (int i = 0; i < 4; ++i) {
    v8f S[4];
#pragma unroll
    for (int j = 0; j < 4; ++j) S[j] = wmma_f16(aq[i], bkf[j], vzero8());

#pragma unroll
    for (int m = 0; m < 8; ++m) {
      int row = 16 * i + m + 8 * g;
      float vals[4];
      float mx = -1e30f;
#pragma unroll
      for (int j = 0; j < 4; ++j) {
        int col = 16 * j + n;
        unsigned short pu = pairIdx[row * 64 + col];
        float b = rpeS[(pu & 255) * NHEAD + w] + rpeS[(pu >> 8) * NHEAD + w];
        float s = S[j][m] * scale + b;
        vals[j] = s;
        mx = fmaxf(mx, s);
      }
      // row reductions across the 16 lanes of each half-wave (wave32-safe)
#pragma unroll
      for (int d = 1; d < 16; d <<= 1) mx = fmaxf(mx, __shfl_xor(mx, d, 32));
      float sum = 0.0f;
#pragma unroll
      for (int j = 0; j < 4; ++j) {
        float e = __expf(vals[j] - mx);
        vals[j] = e;
        sum += e;
      }
#pragma unroll
      for (int d = 1; d < 16; d <<= 1) sum += __shfl_xor(sum, d, 32);
      float inv = 1.0f / sum;
#pragma unroll
      for (int j = 0; j < 4; ++j)
        probW[(m + 8 * g) * PRS + 16 * j + n] = (_Float16)(vals[j] * inv);
    }

    // wave-private round trip: read probs back in A-frag layout
    v16h ap0 = load_a_frag(probW, 0, PRS, 0, lane);
    v16h ap1 = load_a_frag(probW, 0, PRS, 32, lane);
#pragma unroll
    for (int jj = 0; jj < 2; ++jj) {
      v8f o = wmma_f16(ap0, bvf[0][jj], vzero8());
      o = wmma_f16(ap1, bvf[1][jj], o);
      int col = w * DHEAD + 16 * jj + n;
#pragma unroll
      for (int m = 0; m < 8; ++m) {
        int token = 16 * i + m + 8 * g;
        Attn[token * XS + col] = (_Float16)o[m];
      }
    }
  }
  __syncthreads();

  // ---------------- Phase 5: output projection ----------------
  v8f oacc[4][2];
#pragma unroll
  for (int i = 0; i < 4; ++i) {
    oacc[i][0] = vzero8();
    oacc[i][1] = vzero8();
  }
  for (int k0 = 0; k0 < DMODEL; k0 += 32) {
    stage_wtile(wo, k0, Wt, tid);
    if (k0 + 32 < DMODEL)
      __builtin_prefetch(wo + (k0 + 32) * DMODEL + tid * 32, 0, 1);
    __syncthreads();
    v16h bf0 = load_b_frag(Wt, 32 * w, WS, 0, lane);
    v16h bf1 = load_b_frag(Wt, 32 * w + 16, WS, 0, lane);
#pragma unroll
    for (int i = 0; i < 4; ++i) {
      v16h af = load_a_frag(Attn, 16 * i, XS, k0, lane);
      oacc[i][0] = wmma_f16(af, bf0, oacc[i][0]);
      oacc[i][1] = wmma_f16(af, bf1, oacc[i][1]);
    }
    __syncthreads();
  }
#pragma unroll
  for (int jj = 0; jj < 2; ++jj) {
    int col = 32 * w + 16 * jj + n;
    float bias = bo[col];
#pragma unroll
    for (int i = 0; i < 4; ++i) {
#pragma unroll
      for (int m = 0; m < 8; ++m) {
        int token = 16 * i + m + 8 * g;
        out[(pbase + token) * DMODEL + col] = oacc[i][jj][m] + bias;
      }
    }
  }
}

extern "C" void kernel_launch(void* const* d_in, const int* in_sizes, int n_in,
                              void* d_out, int out_size, void* d_ws, size_t ws_size,
                              hipStream_t stream) {
  (void)in_sizes; (void)n_in; (void)out_size; (void)ws_size;
  const float* x      = (const float*)d_in[0];
  const float* coords = (const float*)d_in[1];
  const float* wq     = (const float*)d_in[2];
  const float* bq     = (const float*)d_in[3];
  const float* wk     = (const float*)d_in[4];
  const float* bk     = (const float*)d_in[5];
  const float* wv     = (const float*)d_in[6];
  const float* bv     = (const float*)d_in[7];
  const float* wo     = (const float*)d_in[8];
  const float* bo     = (const float*)d_in[9];
  const float* rpe    = (const float*)d_in[10];
  float* out = (float*)d_out;

  hipMemsetAsync(d_ws, 0, sizeof(unsigned int), stream);
  eta_range_kernel<<<NPATCH, 64, 0, stream>>>(coords, (unsigned int*)d_ws);
  patched_attn_kernel<<<NPATCH, 256, 0, stream>>>(
      x, coords, wq, bq, wk, bk, wv, bv, wo, bo, rpe, out, (const float*)d_ws);
}
